// PNANet_28432683499892
// MI455X (gfx1250) — matrix operand order repaired and checked by reading, not compiled, not run on hardware
//
#include <hip/hip_runtime.h>
#include <math.h>

// ---------- dims (from reference) ----------
#define T_TOW   4
#define F_DIM   32          // H == F == ED
#define KCOMB   96          // 3F combined input rows
#define CW      128         // T*F message columns
#define FO_DIM  8
#define NGRAPH  64
#define EPSV    1e-5f

typedef __attribute__((ext_vector_type(16))) __bf16 v16bf;
typedef __attribute__((ext_vector_type(8)))  float  v8f;

union Frag { v16bf v; unsigned int d[8]; unsigned short u[16]; };

// ---------- optional TDM (Tensor Data Mover) support ----------
#if defined(__has_builtin)
#if __has_builtin(__builtin_amdgcn_tensor_load_to_lds) && __has_builtin(__builtin_amdgcn_s_wait_tensorcnt)
#define USE_TDM 1
#endif
#endif

#ifdef USE_TDM
typedef unsigned int u32x4 __attribute__((ext_vector_type(4)));
typedef int i32x4 __attribute__((ext_vector_type(4)));
typedef int i32x8 __attribute__((ext_vector_type(8)));

// 1-D contiguous DMA: global -> LDS, elem4 = number of 4-byte elements (< 65536).
// D# group0: count=1 | lds_addr | global_addr[56:0] | type=2
// D# group1: data_size=4B, tensor_dim0=tile_dim0=elem4, tensor_dim1=tile_dim1=1,
//            tensor_dim0_stride=elem4, no pad / no multicast / no barrier.
__device__ __forceinline__ void tdm_load_1d(const void* gptr, void* lptr, unsigned elem4) {
    unsigned long long ga = (unsigned long long)gptr;
    unsigned int lds = (unsigned int)(unsigned long long)lptr;  // LDS aperture: low 32 bits
    u32x4 g0;
    g0[0] = 1u;                                                   // count=1 (valid, user)
    g0[1] = lds;                                                  // lds_addr
    g0[2] = (unsigned int)ga;                                     // global_addr[31:0]
    g0[3] = (unsigned int)((ga >> 32) & 0x1FFFFFFu) | (2u << 30); // addr[56:32] | type=2
    i32x8 g1;
    g1[0] = (int)(2u << 16);                        // data_size = 4 bytes
    g1[1] = (int)((elem4 & 0xFFFFu) << 16);         // tensor_dim0[15:0]
    g1[2] = (int)((elem4 >> 16) & 0xFFFFu) | (1 << 16); // tensor_dim0[31:16] | tensor_dim1=1
    g1[3] = (int)((elem4 & 0xFFFFu) << 16);         // tile_dim0
    g1[4] = 1;                                      // tile_dim1 = 1
    g1[5] = (int)elem4;                             // tensor_dim0_stride[31:0]
    g1[6] = 0;
    g1[7] = 0;
    i32x4 z4 = {0, 0, 0, 0};
#if __clang_major__ >= 23
    i32x8 z8 = {0, 0, 0, 0, 0, 0, 0, 0};
    __builtin_amdgcn_tensor_load_to_lds(g0, g1, z4, z4, z8, 0);
#else
    __builtin_amdgcn_tensor_load_to_lds(g0, g1, z4, z4, 0);
#endif
}
#endif

// ---------- helpers ----------
__device__ __forceinline__ unsigned short f2bf(float f) {
    unsigned int u = __float_as_uint(f);
    unsigned int r = (u + 0x7FFFu + ((u >> 16) & 1u)) >> 16;
    return (unsigned short)r;
}
__device__ __forceinline__ unsigned int pack2bf(float a, float b) {
    return (unsigned int)f2bf(a) | ((unsigned int)f2bf(b) << 16);
}
__device__ __forceinline__ void atomicMaxF(float* a, float v) {
    if (v >= 0.f) atomicMax((int*)a, __float_as_int(v));
    else          atomicMin((unsigned int*)a, __float_as_uint(v));
}
__device__ __forceinline__ void atomicMinF(float* a, float v) {
    if (v >= 0.f) atomicMin((int*)a, __float_as_int(v));
    else          atomicMax((unsigned int*)a, __float_as_uint(v));
}

// A-matrix (16-bit, 16xK tile) lane layout: lane m=lane&15, half=lane>>4.
// VGPR v packs K,K+1 with K = (v<4 ? 2v : 16+2(v-4)) + 8*half + kbase.
__device__ __forceinline__ void load_a_frag(Frag& f, const unsigned short* row0,
                                            int stride, int lane, int kbase) {
    int m = lane & 15, half = lane >> 4;
    const unsigned short* r = row0 + m * stride;
#pragma unroll
    for (int v = 0; v < 8; ++v) {
        int k = ((v < 4) ? (2 * v) : (16 + 2 * (v - 4))) + 8 * half + kbase;
        f.d[v] = *(const unsigned int*)(r + k);
    }
}
// B-matrix (16-bit, Kx16 tile), weights pre-packed as uint pairs Wp[kp][ncols]:
// lane n=lane&15, half=lane>>4, VGPR v: kp = (v<4 ? v : v+4) + 4*half + kpbase.
__device__ __forceinline__ void load_b_frag(Frag& f, const unsigned int* Wp,
                                            int ncols, int colbase, int lane, int kpbase) {
    int n = lane & 15, half = lane >> 4;
#pragma unroll
    for (int v = 0; v < 8; ++v) {
        int kp = ((v < 4) ? v : (v + 4)) + 4 * half + kpbase;
        f.d[v] = Wp[kp * ncols + colbase + n];
    }
}
__device__ __forceinline__ v8f wmma_bf16(const Frag& a, const Frag& b, v8f c) {
    return __builtin_amdgcn_wmma_f32_16x16x32_bf16(false, a.v, false, b.v,
                                                   (short)0, c, false, false);
}

// ---------- setup kernels ----------
__global__ void k_init0(float* deg, float* gpool, float* sumlog, int N) {
    int i = blockIdx.x * 256 + threadIdx.x;
    if (i < N) deg[i] = 0.f;
    if (i < NGRAPH * F_DIM) gpool[i] = 0.f;
    if (i == 0) sumlog[0] = 0.f;
}
__global__ void k_deg(const int* __restrict__ ei, float* deg, int E) {
    int i = blockIdx.x * 256 + threadIdx.x;
    if (i < E) atomicAdd(deg + ei[(size_t)E + i], 1.f);
}
__global__ void k_sumlog(const float* __restrict__ deg, float* sumlog, int N) {
    __shared__ float sr[256];
    int i = blockIdx.x * 256 + threadIdx.x;
    sr[threadIdx.x] = (i < N) ? logf(deg[i] + 1.f) : 0.f;
    __syncthreads();
    for (int o = 128; o > 0; o >>= 1) {
        if (threadIdx.x < o) sr[threadIdx.x] += sr[threadIdx.x + o];
        __syncthreads();
    }
    if (threadIdx.x == 0) atomicAdd(sumlog, sr[0]);
}
__global__ void k_scalers(const float* __restrict__ deg, const float* __restrict__ sumlog,
                          float* amp, float* att, int N) {
    int i = blockIdx.x * 256 + threadIdx.x;
    if (i >= N) return;
    float avg = sumlog[0] / (float)N;
    float degc = fmaxf(deg[i], 1.f);
    float ld = logf(degc + 1.f);
    amp[i] = ld / avg;
    att[i] = avg / ld;
}
__global__ void k_cvt_x(const float* __restrict__ x, unsigned short* xb, long long total) {
    long long i = (long long)blockIdx.x * 256 + threadIdx.x;
    if (i < total) xb[i] = f2bf(x[i]);
}
// pack node_W [128,32] into uint pairs [64][32]
__global__ void k_cvt_nodew(const float* __restrict__ W, unsigned int* Wp) {
    int i = blockIdx.x * 256 + threadIdx.x;
    if (i >= 64 * 32) return;
    int kp = i >> 5, c = i & 31;
    Wp[i] = pack2bf(W[(2 * kp) * 32 + c], W[(2 * kp + 1) * 32 + c]);
}
// e = edge_attr @ edge_W + edge_b  -> bf16  (K=16, small VALU GEMM)
__global__ void k_edge_emb(const float* __restrict__ ea, const float* __restrict__ eW,
                           const float* __restrict__ eb, unsigned short* e_bf, int E) {
    __shared__ float sw[16 * 32];
    __shared__ float sb[32];
    int t = threadIdx.x;
    if (t < 512) sw[t] = eW[t];
    if (t < 32) sb[t] = eb[t];
    __syncthreads();
    long long idx = (long long)blockIdx.x * 256 + t;
    if (idx >= (long long)E * 32) return;
    int e = (int)(idx >> 5), c = (int)(idx & 31);
    const float* row = ea + (size_t)e * 16;
    float acc = sb[c];
#pragma unroll
    for (int j = 0; j < 16; ++j) acc += row[j] * sw[j * 32 + c];
    e_bf[idx] = f2bf(acc);
}

// ---------- node embedding: h = x @ node_W + b  (bf16 WMMA, M-tile 16, K=128) ----------
__global__ void k_node_emb(const unsigned short* __restrict__ x_bf,
                           const unsigned int* __restrict__ nWp,
                           const float* __restrict__ node_b,
                           float* h, unsigned short* h_bf, int N) {
    __shared__ alignas(16) unsigned short sX[128 * 128];
    __shared__ alignas(16) unsigned int   sW[64 * 32];
    int tid = threadIdx.x;
    int nb = blockIdx.x * 128;
    bool full = (nb + 128 <= N);
#ifdef USE_TDM
    if (full) {
        // wave 0 issues both DMAs; other waves just hit the barrier
        if (tid < 32) {
            tdm_load_1d(nWp, sW, 64 * 32);                         // 8 KB weights
            tdm_load_1d(x_bf + (size_t)nb * 128, sX, 128 * 128 / 2); // 32 KB x tile
            __builtin_amdgcn_s_wait_tensorcnt(0);
        }
    } else
#endif
    {
        for (int i = tid; i < 512; i += 256)
            ((uint4*)sW)[i] = ((const uint4*)nWp)[i];
        for (int i = tid; i < 128 * 16; i += 256) {
            int r = i >> 4, q = i & 15;
            int node = nb + r; if (node >= N) node = N - 1;
            ((uint4*)(sX + (size_t)r * 128))[q] =
                ((const uint4*)(x_bf + (size_t)node * 128))[q];
        }
    }
    __syncthreads();
    int wv = tid >> 5, lane = tid & 31, half = lane >> 4;
    Frag a[4];
#pragma unroll
    for (int ks = 0; ks < 4; ++ks) load_a_frag(a[ks], sX + wv * 16 * 128, 128, lane, ks * 32);
#pragma unroll
    for (int nt = 0; nt < 2; ++nt) {
        v8f c = {};
#pragma unroll
        for (int ks = 0; ks < 4; ++ks) {
            Frag b; load_b_frag(b, sW, 32, nt * 16, lane, ks * 16);
            c = wmma_bf16(a[ks], b, c);
        }
        int col = nt * 16 + (lane & 15);
        float bias = node_b[col];
        if (full) {
#pragma unroll
            for (int v = 0; v < 8; ++v) {
                int node = nb + wv * 16 + v + 8 * half;
                float val = c[v] + bias;
                h[(size_t)node * 32 + col] = val;
                h_bf[(size_t)node * 32 + col] = f2bf(val);
            }
        } else {
#pragma unroll
            for (int v = 0; v < 8; ++v) {
                int node = nb + wv * 16 + v + 8 * half;
                if (node < N) {
                    float val = c[v] + bias;
                    h[(size_t)node * 32 + col] = val;
                    h_bf[(size_t)node * 32 + col] = f2bf(val);
                }
            }
        }
    }
}

// ---------- per-layer combined weight prep ----------
// Wcomb[96][128]: rows 0..63 = pre_W[l] (dst/src parts), rows 64..95 = enc_W[l] @ pre_W_e.
// Stored pair-packed bf16: Wp[kp*128+col] = {W[2kp][col], W[2kp+1][col]}.
// bc[col] = pre_b + enc_b @ pre_W_e.
__global__ void k_prep(const float* __restrict__ preW, const float* __restrict__ preB,
                       const float* __restrict__ encW, const float* __restrict__ encB,
                       int l, unsigned int* Wp, float* bc) {
    int idx = blockIdx.x * 256 + threadIdx.x;
    if (idx < 48 * 128) {
        int kp = idx >> 7, col = idx & 127;
        int t = col >> 5, g = col & 31;
        const float* PW = preW + (size_t)(l * 4 + t) * 96 * 32;
        float w[2];
#pragma unroll
        for (int q = 0; q < 2; ++q) {
            int f = 2 * kp + q;
            if (f < 64) w[q] = PW[f * 32 + g];
            else {
                int j = f - 64;
                const float* EW = encW + (size_t)l * 32 * 32 + (size_t)j * 32;
                float s = 0.f;
                for (int k = 0; k < 32; ++k) s += EW[k] * PW[(64 + k) * 32 + g];
                w[q] = s;
            }
        }
        Wp[idx] = pack2bf(w[0], w[1]);
    } else if (idx < 48 * 128 + 128) {
        int col = idx - 48 * 128;
        int t = col >> 5, g = col & 31;
        const float* PW = preW + (size_t)(l * 4 + t) * 96 * 32;
        float b = preB[(l * 4 + t) * 32 + g];
        const float* EB = encB + (size_t)l * 32;
        for (int k = 0; k < 32; ++k) b += EB[k] * PW[(64 + k) * 32 + g];
        bc[col] = b;
    }
}
__global__ void k_init_agg(float* s, float* s2, float* mn, float* mx,
                           float* colsum, float* colsumsq, long long total) {
    long long i = (long long)blockIdx.x * 256 + threadIdx.x;
    if (i < total) {
        s[i] = 0.f; s2[i] = 0.f;
        mn[i] = __builtin_inff(); mx[i] = -__builtin_inff();
    }
    if (i < 32) { colsum[i] = 0.f; colsumsq[i] = 0.f; }
}

// ---------- fused edge message GEMM + segment sum/sumsq/min/max ----------
// block: 256 threads (8 wave32), 128 edges; per wave: 16 edges x 128 cols,
// 8 N-tiles x 3 K-steps of v_wmma_f32_16x16x32_bf16 (K=96 total).
// Weights DMA'd via TDM (overlapped with indexed VMEM gathers).
__global__ void k_edge_msg(const int* __restrict__ ei, const float* __restrict__ ew,
                           const unsigned short* __restrict__ h_bf,
                           const unsigned short* __restrict__ e_bf,
                           const unsigned int* __restrict__ Wp, const float* __restrict__ bc,
                           float* s, float* s2, float* mn, float* mx, int E) {
    __shared__ alignas(16) unsigned int   sW[48 * 128];   // 24 KB pair-packed weights
    __shared__ alignas(16) unsigned short sM[128 * 104];  // 26 KB m_in tile (pad 104)
    __shared__ float sEW[128];
    __shared__ int   sDst[128];
    __shared__ float sBc[128];
    int tid = threadIdx.x;
    long long e0 = (long long)blockIdx.x * 128;
    bool full = (e0 + 128 <= (long long)E);
#ifdef USE_TDM
    if (tid < 32) tdm_load_1d(Wp, sW, 48 * 128);   // async: overlaps gathers below
#else
    for (int i = tid; i < 1536; i += 256)
        ((uint4*)sW)[i] = ((const uint4*)Wp)[i];
#endif
    if (tid < 128) sBc[tid] = bc[tid];
    if (tid < 128) {
        long long eid = e0 + tid;
        if (eid < E) { sDst[tid] = ei[(size_t)E + eid]; sEW[tid] = ew[eid]; }
        else         { sDst[tid] = -1;                  sEW[tid] = 0.f; }
    }
    // gather m_in = [h[dst] | h[src] | e]: 3 segs x 128 rows of 32 bf16 (4x uint4)
    for (int r = tid; r < 384; r += 256) {
        int edge = r & 127, seg = r >> 7;
        long long eid = e0 + edge;
        uint4* dst4 = (uint4*)(sM + (size_t)edge * 104 + seg * 32);
        if (eid < E) {
            const unsigned short* src;
            if (seg == 0)      src = h_bf + (size_t)ei[(size_t)E + eid] * 32;
            else if (seg == 1) src = h_bf + (size_t)ei[eid] * 32;
            else               src = e_bf + (size_t)eid * 32;
            const uint4* s4 = (const uint4*)src;
            dst4[0] = s4[0]; dst4[1] = s4[1]; dst4[2] = s4[2]; dst4[3] = s4[3];
        } else {
            uint4 z = {0, 0, 0, 0};
            dst4[0] = z; dst4[1] = z; dst4[2] = z; dst4[3] = z;
        }
    }
#ifdef USE_TDM
    if (tid < 32) __builtin_amdgcn_s_wait_tensorcnt(0);
#endif
    __syncthreads();
    int wv = tid >> 5, lane = tid & 31, half = lane >> 4, n = lane & 15;
    Frag a[3];
#pragma unroll
    for (int ks = 0; ks < 3; ++ks) load_a_frag(a[ks], sM + wv * 16 * 104, 104, lane, ks * 32);
#pragma unroll
    for (int nt = 0; nt < 8; ++nt) {
        v8f c = {};
#pragma unroll
        for (int ks = 0; ks < 3; ++ks) {
            Frag b; load_b_frag(b, sW, 128, nt * 16, lane, ks * 16);
            c = wmma_bf16(a[ks], b, c);
        }
        int col = nt * 16 + n;
        float bias = sBc[col];
        if (full) {
#pragma unroll
            for (int v = 0; v < 8; ++v) {
                int edge = wv * 16 + v + 8 * half;
                int node = sDst[edge];
                float val = (c[v] + bias) * sEW[edge];
                size_t base = (size_t)node * CW + col;
                atomicAdd(s + base, val);
                atomicAdd(s2 + base, val * val);
                atomicMaxF(mx + base, val);
                atomicMinF(mn + base, val);
            }
        } else {
#pragma unroll
            for (int v = 0; v < 8; ++v) {
                int edge = wv * 16 + v + 8 * half;
                int node = sDst[edge];
                if (node >= 0) {
                    float val = (c[v] + bias) * sEW[edge];
                    size_t base = (size_t)node * CW + col;
                    atomicAdd(s + base, val);
                    atomicAdd(s2 + base, val * val);
                    atomicMaxF(mx + base, val);
                    atomicMinF(mn + base, val);
                }
            }
        }
    }
}

// ---------- fused post_nn + lin + BN column stats (8 nodes / block, 1 wave / node) ----------
__global__ void k_post(const float* __restrict__ h, const float* __restrict__ s,
                       const float* __restrict__ s2, const float* __restrict__ mn,
                       const float* __restrict__ mx, const float* __restrict__ deg,
                       const float* __restrict__ amp, const float* __restrict__ att,
                       const float* __restrict__ postW, const float* __restrict__ postB,
                       const float* __restrict__ linW, const float* __restrict__ linB,
                       int l, float* outb, float* colsum, float* colsumsq, int N) {
    __shared__ float sAgg[8][4][4][32];  // [wave][tower][mean/mn/mx/std][c]
    __shared__ float sH[8][32];
    __shared__ float sV[8][32];
    __shared__ float sLW[32 * 32];
    __shared__ float sLB[32];
    __shared__ float sCS[32], sCQ[32];
    int tid = threadIdx.x;
    for (int i = tid; i < 1024; i += 256) sLW[i] = linW[(size_t)l * 1024 + i];
    if (tid < 32) { sLB[tid] = linB[l * 32 + tid]; sCS[tid] = 0.f; sCQ[tid] = 0.f; }
    int wv = tid >> 5, lane = tid & 31;
    int node = blockIdx.x * 8 + wv;
    bool valid = node < N;
    float degc = 1.f, ampv = 0.f, attv = 0.f;
    if (valid) {
        float dg = deg[node];
        degc = fmaxf(dg, 1.f);
        bool hasE = dg > 0.f;
        ampv = amp[node]; attv = att[node];
        sH[wv][lane] = h[(size_t)node * 32 + lane];
#pragma unroll
        for (int q = 0; q < 4; ++q) {
            int col = lane + q * 32;
            size_t base = (size_t)node * CW + col;
            float sv = s[base], qv = s2[base];
            float mean = sv / degc;
            float var = qv / degc - mean * mean;
            float sd = sqrtf(fmaxf(var, 0.f) + EPSV);
            float mnv = hasE ? mn[base] : 0.f;
            float mxv = hasE ? mx[base] : 0.f;
            int t = col >> 5, cc = col & 31;
            sAgg[wv][t][0][cc] = mean; sAgg[wv][t][1][cc] = mnv;
            sAgg[wv][t][2][cc] = mxv;  sAgg[wv][t][3][cc] = sd;
        }
    }
    __syncthreads();
    float acc = 0.f;
    if (valid) {
        int t = lane >> 3, g = lane & 7;
        const float* PW = postW + (size_t)(l * 4 + t) * 416 * 8;
        acc = postB[(l * 4 + t) * 8 + g];
#pragma unroll 8
        for (int f = 0; f < 32; ++f) acc += sH[wv][f] * PW[f * 8 + g];
        for (int which = 0; which < 4; ++which) {
#pragma unroll 8
            for (int cc = 0; cc < 32; ++cc) {
                int f = 32 + which * 32 + cc;
                float a = sAgg[wv][t][which][cc];
                acc += a * (PW[f * 8 + g] + ampv * PW[(f + 128) * 8 + g]
                                          + attv * PW[(f + 256) * 8 + g]);
            }
        }
    }
    sV[wv][lane] = acc;
    __syncthreads();
    if (valid) {
        float acc2 = sLB[lane];
#pragma unroll 8
        for (int j = 0; j < 32; ++j) acc2 += sV[wv][j] * sLW[j * 32 + lane];
        outb[(size_t)node * 32 + lane] = acc2;
        atomicAdd(&sCS[lane], acc2);
        atomicAdd(&sCQ[lane], acc2 * acc2);
    }
    __syncthreads();
    if (tid < 32) { atomicAdd(colsum + tid, sCS[tid]); atomicAdd(colsumsq + tid, sCQ[tid]); }
}

__global__ void k_bnstats(const float* colsum, const float* colsumsq,
                          const float* bng, const float* bnb, int l, float* stats, int N) {
    int k = threadIdx.x;
    if (k < 32) {
        float mu = colsum[k] / (float)N;
        float var = colsumsq[k] / (float)N - mu * mu;
        float inv = rsqrtf(fmaxf(var, 0.f) + EPSV);
        stats[k] = mu;
        stats[32 + k] = inv * bng[l * 32 + k];
        stats[64 + k] = bnb[l * 32 + k];
    }
}
__global__ void k_bn_relu(const float* __restrict__ outb, const float* __restrict__ stats,
                          float* h, unsigned short* h_bf, int total) {
    int i = blockIdx.x * 256 + threadIdx.x;
    if (i >= total) return;
    int k = i & 31;
    float v = (outb[i] - stats[k]) * stats[32 + k] + stats[64 + k];
    v = fmaxf(v, 0.f);
    h[i] = v; h_bf[i] = f2bf(v);
}
__global__ void k_pool(const float* __restrict__ h, const int* __restrict__ batch,
                       float* gpool, int N) {
    int i = blockIdx.x * 256 + threadIdx.x;
    if (i >= N * 32) return;
    int n = i >> 5, k = i & 31;
    atomicAdd(gpool + (size_t)batch[n] * 32 + k, h[i]);
}
__global__ void k_mlp(const float* __restrict__ g, const float* W1, const float* b1,
                      const float* W2, const float* b2, const float* W3, const float* b3,
                      float* out) {
    int r = threadIdx.x;
    if (r >= NGRAPH) return;
    float gi[32], g1[32], g2[16];
    for (int j = 0; j < 32; ++j) gi[j] = g[r * 32 + j];
    for (int k = 0; k < 32; ++k) {
        float a = b1[k];
        for (int j = 0; j < 32; ++j) a += gi[j] * W1[j * 32 + k];
        g1[k] = fmaxf(a, 0.f);
    }
    for (int k = 0; k < 16; ++k) {
        float a = b2[k];
        for (int j = 0; j < 32; ++j) a += g1[j] * W2[j * 16 + k];
        g2[k] = fmaxf(a, 0.f);
    }
    float a = b3[0];
    for (int j = 0; j < 16; ++j) a += g2[j] * W3[j];
    out[r] = a;
}

// ---------- host ----------
extern "C" void kernel_launch(void* const* d_in, const int* in_sizes, int n_in,
                              void* d_out, int out_size, void* d_ws, size_t ws_size,
                              hipStream_t stream) {
    const float* x        = (const float*)d_in[0];
    const float* ea       = (const float*)d_in[1];
    const float* ew       = (const float*)d_in[2];
    const int*   ei       = (const int*)d_in[3];
    const int*   batch    = (const int*)d_in[4];
    const float* node_W   = (const float*)d_in[5];
    const float* node_b   = (const float*)d_in[6];
    const float* edge_W   = (const float*)d_in[7];
    const float* edge_b   = (const float*)d_in[8];
    const float* enc_W    = (const float*)d_in[9];
    const float* enc_b    = (const float*)d_in[10];
    const float* pre_W    = (const float*)d_in[11];
    const float* pre_b    = (const float*)d_in[12];
    const float* post_W   = (const float*)d_in[13];
    const float* post_b   = (const float*)d_in[14];
    const float* lin_W    = (const float*)d_in[15];
    const float* lin_b    = (const float*)d_in[16];
    const float* bn_g     = (const float*)d_in[17];
    const float* bn_b     = (const float*)d_in[18];
    const float* mlp_W1   = (const float*)d_in[19];
    const float* mlp_b1   = (const float*)d_in[20];
    const float* mlp_W2   = (const float*)d_in[21];
    const float* mlp_b2   = (const float*)d_in[22];
    const float* mlp_W3   = (const float*)d_in[23];
    const float* mlp_b3   = (const float*)d_in[24];

    const int N = in_sizes[4];   // batch
    const int E = in_sizes[2];   // edge_weight

    char* p = (char*)d_ws;
    auto alloc = [&](size_t bytes) -> void* {
        void* r = (void*)p;
        p += (bytes + 255) & ~(size_t)255;
        return r;
    };
    float* deg    = (float*)alloc((size_t)N * 4);
    float* amp    = (float*)alloc((size_t)N * 4);
    float* att    = (float*)alloc((size_t)N * 4);
    float* sumlog = (float*)alloc(256);
    unsigned short* x_bf = (unsigned short*)alloc((size_t)N * 128 * 2);
    unsigned int*   nWp  = (unsigned int*)alloc(64 * 32 * 4);
    float*          hbuf = (float*)alloc((size_t)N * 32 * 4);
    unsigned short* h_bf = (unsigned short*)alloc((size_t)N * 32 * 2);
    unsigned short* e_bf = (unsigned short*)alloc((size_t)E * 32 * 2);
    unsigned int*   Wp   = (unsigned int*)alloc(48 * 128 * 4);
    float*          bc   = (float*)alloc(128 * 4);
    float* sbuf  = (float*)alloc((size_t)N * CW * 4);
    float* s2buf = (float*)alloc((size_t)N * CW * 4);
    float* mnbuf = (float*)alloc((size_t)N * CW * 4);
    float* mxbuf = (float*)alloc((size_t)N * CW * 4);
    float* outb  = (float*)alloc((size_t)N * 32 * 4);
    float* colsum   = (float*)alloc(32 * 4);
    float* colsumsq = (float*)alloc(32 * 4);
    float* stats    = (float*)alloc(96 * 4);
    float* gpool    = (float*)alloc(NGRAPH * 32 * 4);

    auto cdiv = [](long long a, long long b) { return (unsigned)((a + b - 1) / b); };

    k_init0<<<cdiv(N > 2048 ? N : 2048, 256), 256, 0, stream>>>(deg, gpool, sumlog, N);
    k_deg<<<cdiv(E, 256), 256, 0, stream>>>(ei, deg, E);
    k_sumlog<<<cdiv(N, 256), 256, 0, stream>>>(deg, sumlog, N);
    k_scalers<<<cdiv(N, 256), 256, 0, stream>>>(deg, sumlog, amp, att, N);
    k_cvt_x<<<cdiv((long long)N * 128, 256), 256, 0, stream>>>(x, x_bf, (long long)N * 128);
    k_cvt_nodew<<<cdiv(64 * 32, 256), 256, 0, stream>>>(node_W, nWp);
    k_edge_emb<<<cdiv((long long)E * 32, 256), 256, 0, stream>>>(ea, edge_W, edge_b, e_bf, E);
    k_node_emb<<<cdiv(N, 128), 256, 0, stream>>>(x_bf, nWp, node_b, hbuf, h_bf, N);

    for (int l = 0; l < 2; ++l) {
        k_prep<<<cdiv(48 * 128 + 128, 256), 256, 0, stream>>>(pre_W, pre_b, enc_W, enc_b, l, Wp, bc);
        k_init_agg<<<cdiv((long long)N * CW, 256), 256, 0, stream>>>(
            sbuf, s2buf, mnbuf, mxbuf, colsum, colsumsq, (long long)N * CW);
        k_edge_msg<<<cdiv(E, 128), 256, 0, stream>>>(ei, ew, h_bf, e_bf, Wp, bc,
                                                     sbuf, s2buf, mnbuf, mxbuf, E);
        k_post<<<cdiv(N, 8), 256, 0, stream>>>(hbuf, sbuf, s2buf, mnbuf, mxbuf, deg, amp, att,
                                               post_W, post_b, lin_W, lin_b, l,
                                               outb, colsum, colsumsq, N);
        k_bnstats<<<1, 32, 0, stream>>>(colsum, colsumsq, bn_g, bn_b, l, stats, N);
        k_bn_relu<<<cdiv((long long)N * 32, 256), 256, 0, stream>>>(outb, stats, hbuf, h_bf, N * 32);
    }
    k_pool<<<cdiv((long long)N * 32, 256), 256, 0, stream>>>(hbuf, batch, gpool, N);
    k_mlp<<<1, 64, 0, stream>>>(gpool, mlp_W1, mlp_b1, mlp_W2, mlp_b2, mlp_W3, mlp_b3,
                                (float*)d_out);
}
